// EGNNLayer_53979148976481
// MI455X (gfx1250) — compile-verified, matrix operand order checked by reading, not run on hardware
//
#include <hip/hip_runtime.h>
#include <hip/hip_bf16.h>

typedef __attribute__((ext_vector_type(16))) _Float16 v16h;
typedef __attribute__((ext_vector_type(8)))  float    v8f;

#define NNODE 50000
#define NEDGE 800000
#define IND   64
#define HID   128
#define EDD   16
#define KE    160   // 2*IN + 1 + ED = 145, padded to 160 (5 K-chunks of 32)
#define KN    192   // IN + H (6 K-chunks)

// wave-local LDS fence: DScnt is per-wave, lanes issue in lockstep
#define LDS_SYNC() asm volatile("s_wait_dscnt 0" ::: "memory")

// fast SiLU: v * rcp(1+exp(-v)) -> v_exp_f32 + v_add + v_rcp_f32 + v_mul
__device__ __forceinline__ float silu(float v) {
    return v * __builtin_amdgcn_rcpf(1.f + __expf(-v));
}

// pack two f32 into two f16 in one dword
__device__ __forceinline__ unsigned pk2h(float a, float b) {
    union { _Float16 h[2]; unsigned u; } t;
    t.h[0] = (_Float16)a; t.h[1] = (_Float16)b;
    return t.u;
}

// A-fragment (16x32 f16): lane l holds row m=l&15; half-pair j at
// k = (j<4 ? 2j : 16+2(j-4)) + 8*(l>>4). Gather 8 aligned half-pairs from LDS.
__device__ __forceinline__ v16h load_a_frag(const _Float16* base, int lane, int kbase, int stride) {
    int m = lane & 15, hs = lane >> 4;
    union { v16h v; unsigned u[8]; } f;
    const _Float16* p = base + m * stride + kbase + 8 * hs;
#pragma unroll
    for (int r = 0; r < 8; ++r) {
        int k0 = (r < 4) ? (2 * r) : (16 + 2 * (r - 4));
        f.u[r] = *(const unsigned*)(p + k0);
    }
    return f.v;
}

// Repack a [K,N] row-major fp32 weight matrix into per-fragment f16 B operands.
__global__ void pack_weights(const float* __restrict__ W, _Float16* __restrict__ out,
                             int K, int N, int NT) {
    int frag = blockIdx.x;
    int nt = frag % NT;
    int kc = frag / NT;
    int lane = threadIdx.x;
    int nC = lane & 15, hs = lane >> 4;
    int n = nt * 16 + nC;
#pragma unroll
    for (int j = 0; j < 16; ++j) {
        int r = j >> 1, p = j & 1;
        int kl = ((r < 4) ? 2 * r : 16 + 2 * (r - 4)) + p + 8 * hs;
        int k = kc * 32 + kl;
        float v = (k < K) ? W[(long)k * N + n] : 0.f;
        out[((long)frag * 32 + lane) * 16 + j] = (_Float16)v;
    }
}

// ---------------- edge MLP + scatters: 1 wave = 16 edges ----------------
__global__ void __launch_bounds__(128) edge_kernel(
    const float* __restrict__ x, const float* __restrict__ pos,
    const int* __restrict__ ei, const float* __restrict__ ea,
    const float* __restrict__ s,
    const _Float16* __restrict__ We1f, const float* __restrict__ be1,
    const _Float16* __restrict__ We2f, const float* __restrict__ be2,
    const float* __restrict__ Wc, const float* __restrict__ bc,
    float* __restrict__ m_sum, float* __restrict__ deg, float* __restrict__ coord)
{
    __shared__ _Float16 ein_s[4][16][KE];
    __shared__ _Float16 h_s[4][16][HID];
    __shared__ float sg_s[4][16];
    __shared__ float df_s[4][16][3];
    __shared__ int   di_s[4][16];
    __shared__ int   si_s[4][16];

    int w = threadIdx.x >> 5, lane = threadIdx.x & 31;
    int e0 = (blockIdx.x * 4 + w) * 16;
    int nC = lane & 15, hs = lane >> 4;

    if (lane < 16) {
        int e = e0 + lane;
        int si = ei[e];          // edge_index[0] = src
        int di = ei[NEDGE + e];  // edge_index[1] = dst
        si_s[w][lane] = si;
        di_s[w][lane] = di;
        sg_s[w][lane] = s[si];
        float dx = pos[di * 3 + 0] - pos[si * 3 + 0];
        float dy = pos[di * 3 + 1] - pos[si * 3 + 1];
        float dz = pos[di * 3 + 2] - pos[si * 3 + 2];
        df_s[w][lane][0] = dx; df_s[w][lane][1] = dy; df_s[w][lane][2] = dz;
        ein_s[w][lane][2 * IND] = (_Float16)(dx * dx + dy * dy + dz * dz); // r2 @ col 128
#pragma unroll
        for (int c = 2 * IND + 1 + EDD; c < KE; ++c)  // pad cols 145..159
            ein_s[w][lane][c] = (_Float16)0.f;
    }
    LDS_SYNC();

    // stage x[dst] -> cols [0,64), x[src] -> [64,128): float4 loads, b64 LDS stores
    for (int idx = lane; idx < 16 * (IND / 4); idx += 32) {
        int m = idx >> 4, c4 = (idx & 15) * 4;
        float4 vd = *(const float4*)(x + (long)di_s[w][m] * IND + c4);
        float4 vs = *(const float4*)(x + (long)si_s[w][m] * IND + c4);
        *(uint2*)&ein_s[w][m][c4]       = make_uint2(pk2h(vd.x, vd.y), pk2h(vd.z, vd.w));
        *(uint2*)&ein_s[w][m][IND + c4] = make_uint2(pk2h(vs.x, vs.y), pk2h(vs.z, vs.w));
    }
    // edge_attr -> cols [129,145) (odd start: scalar stores)
    for (int idx = lane; idx < 16 * EDD; idx += 32) {
        int m = idx >> 4, c = idx & 15;
        ein_s[w][m][2 * IND + 1 + c] = (_Float16)ea[(long)(e0 + m) * EDD + c];
    }
    LDS_SYNC();

    // GEMM1: [16,160] x [160,128], SiLU -> h_s (restage C-layout -> A-layout)
    for (int nt = 0; nt < 8; ++nt) {
        float b = be1[nt * 16 + nC];
        v8f c = {b, b, b, b, b, b, b, b};
#pragma unroll
        for (int kc = 0; kc < 5; ++kc) {
            v16h a  = load_a_frag(&ein_s[w][0][0], lane, kc * 32, KE);
            v16h bm = *(const v16h*)(We1f + ((long)(kc * 8 + nt) * 32 + lane) * 16);
            c = __builtin_amdgcn_wmma_f32_16x16x32_f16(false, a, false, bm, (short)0, c, false, false);
        }
#pragma unroll
        for (int r = 0; r < 8; ++r)
            h_s[w][r + 8 * hs][nt * 16 + nC] = (_Float16)silu(c[r]);
    }
    LDS_SYNC();

    // GEMM2: [16,128] x [128,128], SiLU, gate, scatter + gamma partials
    float gp[8] = {0, 0, 0, 0, 0, 0, 0, 0};
    for (int nt = 0; nt < 8; ++nt) {
        int n = nt * 16 + nC;
        float b = be2[n];
        v8f c = {b, b, b, b, b, b, b, b};
#pragma unroll
        for (int kc = 0; kc < 4; ++kc) {
            v16h a  = load_a_frag(&h_s[w][0][0], lane, kc * 32, HID);
            v16h bm = *(const v16h*)(We2f + ((long)(kc * 8 + nt) * 32 + lane) * 16);
            c = __builtin_amdgcn_wmma_f32_16x16x32_f16(false, a, false, bm, (short)0, c, false, false);
        }
        float wc = Wc[n];
#pragma unroll
        for (int r = 0; r < 8; ++r) {
            int m = r + 8 * hs;
            float v = silu(c[r]) * sg_s[w][m];   // gate by s[src]
            gp[r] += v * wc;                      // gamma partial
            atomicAdd(&m_sum[(long)di_s[w][m] * HID + n], v);
        }
    }
    // reduce gamma over the 16 lanes of each half-wave (hs groups stay disjoint)
#pragma unroll
    for (int off = 1; off < 16; off <<= 1)
#pragma unroll
        for (int r = 0; r < 8; ++r) gp[r] += __shfl_xor(gp[r], off, 32);
    if (nC == 0) {
        float bcv = bc[0];
#pragma unroll
        for (int r = 0; r < 8; ++r) {
            int m = r + 8 * hs;
            int d = di_s[w][m];
            float g = gp[r] + bcv;
            atomicAdd(&coord[d * 3 + 0], g * df_s[w][m][0]);
            atomicAdd(&coord[d * 3 + 1], g * df_s[w][m][1]);
            atomicAdd(&coord[d * 3 + 2], g * df_s[w][m][2]);
            atomicAdd(&deg[d], 1.0f);
        }
    }
}

// ---------------- node MLP: 1 wave = 16 nodes ----------------
__global__ void __launch_bounds__(128) node_kernel(
    const float* __restrict__ x, const float* __restrict__ m_sum,
    const float* __restrict__ deg,
    const _Float16* __restrict__ Wn1f, const float* __restrict__ bn1,
    const _Float16* __restrict__ Wn2f, const float* __restrict__ bn2,
    float* __restrict__ out_x)
{
    __shared__ _Float16 ain_s[4][16][KN];
    __shared__ _Float16 h_s[4][16][HID];
    __shared__ float rdeg_s[4][16];

    int w = threadIdx.x >> 5, lane = threadIdx.x & 31;
    int i0 = (blockIdx.x * 4 + w) * 16;
    if (i0 >= NNODE) return;
    int nC = lane & 15, hs = lane >> 4;

    if (lane < 16) {
        int i = i0 + lane; if (i >= NNODE) i = NNODE - 1;
        float dg = deg[i]; if (dg < 1.f) dg = 1.f;
        rdeg_s[w][lane] = __builtin_amdgcn_rcpf(dg);
    }
    LDS_SYNC();
    // x -> cols [0,64): float4 loads
    for (int idx = lane; idx < 16 * (IND / 4); idx += 32) {
        int m = idx >> 4, c4 = (idx & 15) * 4;
        int i = i0 + m; if (i >= NNODE) i = NNODE - 1;
        float4 v = *(const float4*)(x + (long)i * IND + c4);
        *(uint2*)&ain_s[w][m][c4] = make_uint2(pk2h(v.x, v.y), pk2h(v.z, v.w));
    }
    // m_sum/deg -> cols [64,192): float4 loads
    for (int idx = lane; idx < 16 * (HID / 4); idx += 32) {
        int m = idx >> 5, c4 = (idx & 31) * 4;
        int i = i0 + m; if (i >= NNODE) i = NNODE - 1;
        float4 v = *(const float4*)(m_sum + (long)i * HID + c4);
        float r = rdeg_s[w][m];
        *(uint2*)&ain_s[w][m][IND + c4] =
            make_uint2(pk2h(v.x * r, v.y * r), pk2h(v.z * r, v.w * r));
    }
    LDS_SYNC();

    // GEMM1: [16,192] x [192,128], SiLU
    for (int nt = 0; nt < 8; ++nt) {
        float b = bn1[nt * 16 + nC];
        v8f c = {b, b, b, b, b, b, b, b};
#pragma unroll
        for (int kc = 0; kc < 6; ++kc) {
            v16h a  = load_a_frag(&ain_s[w][0][0], lane, kc * 32, KN);
            v16h bm = *(const v16h*)(Wn1f + ((long)(kc * 8 + nt) * 32 + lane) * 16);
            c = __builtin_amdgcn_wmma_f32_16x16x32_f16(false, a, false, bm, (short)0, c, false, false);
        }
#pragma unroll
        for (int r = 0; r < 8; ++r)
            h_s[w][r + 8 * hs][nt * 16 + nC] = (_Float16)silu(c[r]);
    }
    LDS_SYNC();

    // GEMM2: [16,128] x [128,64] -> x_new
    for (int nt = 0; nt < 4; ++nt) {
        int n = nt * 16 + nC;
        float b = bn2[n];
        v8f c = {b, b, b, b, b, b, b, b};
#pragma unroll
        for (int kc = 0; kc < 4; ++kc) {
            v16h a  = load_a_frag(&h_s[w][0][0], lane, kc * 32, HID);
            v16h bm = *(const v16h*)(Wn2f + ((long)(kc * 4 + nt) * 32 + lane) * 16);
            c = __builtin_amdgcn_wmma_f32_16x16x32_f16(false, a, false, bm, (short)0, c, false, false);
        }
#pragma unroll
        for (int r = 0; r < 8; ++r) {
            int i = i0 + r + 8 * hs;
            if (i < NNODE) out_x[(long)i * IND + n] = c[r];
        }
    }
}

__global__ void pos_kernel(const float* __restrict__ pos, const float* __restrict__ coord,
                           const float* __restrict__ deg, float* __restrict__ out_pos) {
    int t = blockIdx.x * blockDim.x + threadIdx.x;
    if (t >= NNODE * 3) return;
    int i = t / 3;
    float dg = deg[i]; if (dg < 1.f) dg = 1.f;
    out_pos[t] = pos[t] + coord[t] * __builtin_amdgcn_rcpf(dg);
}

extern "C" void kernel_launch(void* const* d_in, const int* in_sizes, int n_in,
                              void* d_out, int out_size, void* d_ws, size_t ws_size,
                              hipStream_t stream) {
    const float* x   = (const float*)d_in[0];
    const float* pos = (const float*)d_in[1];
    const int*   ei  = (const int*)d_in[2];
    const float* ea  = (const float*)d_in[3];
    const float* s   = (const float*)d_in[4];
    const float* We1 = (const float*)d_in[5];  const float* be1 = (const float*)d_in[6];
    const float* We2 = (const float*)d_in[7];  const float* be2 = (const float*)d_in[8];
    const float* Wn1 = (const float*)d_in[9];  const float* bn1 = (const float*)d_in[10];
    const float* Wn2 = (const float*)d_in[11]; const float* bn2 = (const float*)d_in[12];
    const float* Wc  = (const float*)d_in[13]; const float* bc  = (const float*)d_in[14];

    float* out_x   = (float*)d_out;              // [N,64]
    float* out_pos = out_x + (long)NNODE * IND;  // [N,3]

    // workspace layout
    float* m_sum = (float*)d_ws;                        // N*128
    float* deg   = m_sum + (long)NNODE * HID;           // N
    float* coord = deg + NNODE;                         // N*3
    _Float16* We1f = (_Float16*)(coord + (long)NNODE * 3);  // 5*8 frags
    _Float16* We2f = We1f + 5 * 8 * 512;                    // 4*8 frags
    _Float16* Wn1f = We2f + 4 * 8 * 512;                    // 6*8 frags
    _Float16* Wn2f = Wn1f + 6 * 8 * 512;                    // 4*4 frags

    size_t acc_bytes = ((long)NNODE * HID + NNODE + (long)NNODE * 3) * sizeof(float);
    hipMemsetAsync(d_ws, 0, acc_bytes, stream);

    pack_weights<<<5 * 8, 32, 0, stream>>>(We1, We1f, 145, 128, 8);
    pack_weights<<<4 * 8, 32, 0, stream>>>(We2, We2f, 128, 128, 8);
    pack_weights<<<6 * 8, 32, 0, stream>>>(Wn1, Wn1f, 192, 128, 8);
    pack_weights<<<4 * 4, 32, 0, stream>>>(Wn2, Wn2f, 128, 64, 4);

    edge_kernel<<<NEDGE / 64, 128, 0, stream>>>(x, pos, ei, ea, s,
                                                We1f, be1, We2f, be2, Wc, bc,
                                                m_sum, deg, coord);
    node_kernel<<<(NNODE + 63) / 64, 128, 0, stream>>>(x, m_sum, deg,
                                                       Wn1f, bn1, Wn2f, bn2, out_x);
    pos_kernel<<<(NNODE * 3 + 255) / 256, 256, 0, stream>>>(pos, coord, deg, out_pos);
}
